// LSTM_56324201120044
// MI455X (gfx1250) — compile-verified, matrix operand order checked by reading, not run on hardware
//
#include <hip/hip_runtime.h>
#include <hip/hip_bf16.h>

// LSTM for MI455X (gfx1250, wave32).
//  Phase 1: x @ [W_ii|W_if|W_ig|W_io] + (b_i*+b_h*) via v_wmma_f32_16x16x32_bf16
//           -> Xpre[t][b][h][gate] f32 (time-major, gate-interleaved).
//  Phase 2: ONE persistent kernel runs all 512 recurrent steps.
//           128 waves = 32 blocks x 4 waves. Wave pair shares a 16-column
//           W_h slice (4 gates x 16 cols x 1024 K, bf16) staged in LDS once
//           (256 KB/block, within the WGP's 320 KB); B-fragments come from
//           LDS every step, A-fragments (h, bf16, double-buffered) from L2.
//           Next step's Xpre slice is prefetched (global_prefetch_b8) while
//           the GEMM runs. Steps separated by an agent-scope atomic
//           generation barrier with s_sleep backoff.

#define BATCH 32
#define SEQ   512
#define IND   1024
#define HIDN  1024
#define NGATE 4
#define SCAN_BLOCKS 32
#define SCAN_THREADS 128     // 4 waves/block -> 128 waves total

typedef unsigned short u16;
typedef __attribute__((ext_vector_type(16))) __bf16 v16bf;
typedef __attribute__((ext_vector_type(8)))  float  v8f;

union FragU { uint4 q[2]; v16bf v; };

__device__ __forceinline__ u16 f2bf(float f) {
  union { float f; unsigned int u; } v; v.f = f;
  unsigned int u = v.u;
  u += 0x7FFFu + ((u >> 16) & 1u);      // round-to-nearest-even
  return (u16)(u >> 16);
}

// 16(M/N) x 32(K) bf16 fragment from row-major memory (stride ld, elements).
// ISA 16-bit A/B layout: lanes 0-15 hold K 0-7/16-23, lanes 16-31 hold
// K 8-15/24-31 of row (lane & 15).
__device__ __forceinline__ v16bf load_frag(const u16* __restrict__ base, int ld) {
  const int lane = threadIdx.x & 31;
  const int row  = lane & 15;
  const int koff = (lane >> 4) << 3;
  const u16* p = base + (size_t)row * ld + koff;
  FragU u;
  u.q[0] = *(const uint4*)(p);
  u.q[1] = *(const uint4*)(p + 16);
  return u.v;
}

__device__ __forceinline__ v8f wmma_bf16(v16bf a, v16bf b, v8f c) {
  return __builtin_amdgcn_wmma_f32_16x16x32_bf16(false, a, false, b,
                                                 (short)0, c, false, false);
}

__device__ __forceinline__ float sigmoidf_(float x) {
  return 1.0f / (1.0f + __expf(-x));
}
__device__ __forceinline__ float tanhf_(float x) {
  float e = __expf(2.0f * x);
  return 1.0f - 2.0f / (e + 1.0f);      // -> -1/+1 at extremes, no NaN
}

// Agent-scope generation barrier across all blocks of the persistent kernel.
__device__ __forceinline__ void grid_sync(unsigned* __restrict__ cnt,
                                          unsigned* __restrict__ gen,
                                          unsigned nblk) {
  __threadfence();                       // publish h/c stores device-wide
  __syncthreads();
  if (threadIdx.x == 0) {
    unsigned g = __hip_atomic_load(gen, __ATOMIC_ACQUIRE, __HIP_MEMORY_SCOPE_AGENT);
    unsigned a = __hip_atomic_fetch_add(cnt, 1u, __ATOMIC_ACQ_REL,
                                        __HIP_MEMORY_SCOPE_AGENT);
    if (a + 1u == nblk) {
      __hip_atomic_store(cnt, 0u, __ATOMIC_RELAXED, __HIP_MEMORY_SCOPE_AGENT);
      __hip_atomic_fetch_add(gen, 1u, __ATOMIC_RELEASE, __HIP_MEMORY_SCOPE_AGENT);
    } else {
      while (__hip_atomic_load(gen, __ATOMIC_ACQUIRE,
                               __HIP_MEMORY_SCOPE_AGENT) == g)
        __builtin_amdgcn_s_sleep(2);
    }
  }
  __syncthreads();
}

// ---------------------------------------------------------------- prep kernels

__global__ void k_cvt_x(const float* __restrict__ x, u16* __restrict__ xbf, int n) {
  int i = blockIdx.x * blockDim.x + threadIdx.x;
  if (i < n) xbf[i] = f2bf(x[i]);
}

// src: f32 [1024 k][1024 n] row-major  ->  dst: bf16 [1024 n][1024 k] row-major
__global__ void k_transpose_cvt(const float* __restrict__ src, u16* __restrict__ dst) {
  __shared__ float tile[32][33];
  const int n0 = blockIdx.x << 5;
  const int k0 = blockIdx.y << 5;
  const int tx = threadIdx.x;           // 0..31
  const int ty = threadIdx.y;           // 0..7
#pragma unroll
  for (int j = 0; j < 32; j += 8)
    tile[ty + j][tx] = src[(size_t)(k0 + ty + j) * HIDN + n0 + tx];
  __syncthreads();
#pragma unroll
  for (int j = 0; j < 32; j += 8)
    dst[(size_t)(n0 + ty + j) * IND + k0 + tx] = f2bf(tile[tx][ty + j]);
}

__global__ void k_bias(const float* __restrict__ bii, const float* __restrict__ bif,
                       const float* __restrict__ big, const float* __restrict__ bio,
                       const float* __restrict__ bhi, const float* __restrict__ bhf,
                       const float* __restrict__ bhg, const float* __restrict__ bho,
                       float* __restrict__ Bias) {
  int i = blockIdx.x * blockDim.x + threadIdx.x;
  if (i >= NGATE * HIDN) return;
  int g = i >> 10, h = i & 1023;
  const float* a = (g == 0) ? bii : (g == 1) ? bif : (g == 2) ? big : bio;
  const float* b = (g == 0) ? bhi : (g == 1) ? bhf : (g == 2) ? bhg : bho;
  Bias[i] = a[h] + b[h];
}

__global__ void k_init(const float* __restrict__ h0, const float* __restrict__ c0,
                       u16* __restrict__ hbf, float* __restrict__ cst,
                       unsigned* __restrict__ bar) {
  int i = blockIdx.x * blockDim.x + threadIdx.x;
  if (i < 2) bar[i] = 0u;               // reset grid-barrier state every call
  if (i >= BATCH * HIDN) return;
  hbf[i] = f2bf(h0[i]);
  cst[i] = c0[i];
}

// ------------------------------------------------ phase 1: input projections
// M=16384 (b*512+s), K=1024, N=4096 (gate*1024+h). Per-wave 32x32 tile.
__global__ __launch_bounds__(256)
void k_gemm_x(const u16* __restrict__ Xbf, const u16* __restrict__ Wt,
              const float* __restrict__ Bias, float* __restrict__ Xpre) {
  const int wid   = (blockIdx.x * blockDim.x + threadIdx.x) >> 5;
  const int ntile = wid & 127;          // 128 tiles of 32 over N=4096
  const int mtile = wid >> 7;           // 512 tiles of 32 over M=16384
  const int m0 = mtile << 5;
  const int n0 = ntile << 5;
  const int g  = n0 >> 10;              // 32-wide tile never crosses a gate
  const int nh = n0 & 1023;

  const u16* A0 = Xbf + (size_t)m0 * IND;
  const u16* A1 = A0 + (size_t)16 * IND;
  const u16* B0 = Wt + ((size_t)g * HIDN + nh) * IND;
  const u16* B1 = B0 + (size_t)16 * IND;

  const v8f z = {0.f, 0.f, 0.f, 0.f, 0.f, 0.f, 0.f, 0.f};
  v8f acc00 = z, acc01 = z, acc10 = z, acc11 = z;

#pragma unroll 4
  for (int k0 = 0; k0 < IND; k0 += 32) {
    v16bf a0 = load_frag(A0 + k0, IND);
    v16bf a1 = load_frag(A1 + k0, IND);
    v16bf b0 = load_frag(B0 + k0, IND);
    v16bf b1 = load_frag(B1 + k0, IND);
    acc00 = wmma_bf16(a0, b0, acc00);
    acc01 = wmma_bf16(a0, b1, acc01);
    acc10 = wmma_bf16(a1, b0, acc10);
    acc11 = wmma_bf16(a1, b1, acc11);
  }

  const int lane = threadIdx.x & 31;
  const int ln = lane & 15;
  const int hi = lane >> 4;
  v8f accs[2][2] = {{acc00, acc01}, {acc10, acc11}};
#pragma unroll
  for (int mi = 0; mi < 2; ++mi)
#pragma unroll
    for (int ni = 0; ni < 2; ++ni) {
      const int n = n0 + ni * 16 + ln;
      const int h = n & 1023;
      const float bias = Bias[g * HIDN + h];
      v8f a = accs[mi][ni];
#pragma unroll
      for (int r = 0; r < 8; ++r) {
        const int Mrow = m0 + mi * 16 + r + (hi << 3);
        const int b = Mrow >> 9;        // batch
        const int s = Mrow & 511;       // time
        Xpre[(((size_t)s * BATCH + b) * HIDN + h) * NGATE + g] = a[r] + bias;
      }
    }
}

// -------------------------------------- phase 2: persistent recurrent scan
// 128 waves (32 blocks x 4). Global wave wid: slice ns = wid>>1 owns hidden
// columns [16ns,16ns+16); m0 = (wid&1)*16 selects the batch half. Each block
// stages its two W_h slices into LDS once; every step then does
// 4 gate WMMAs per k-step with LDS B-fragments and a global bf16 A-fragment.
__global__ __launch_bounds__(SCAN_THREADS)
void k_scan(const u16* __restrict__ Wht, const float* __restrict__ Xpre,
            float* __restrict__ Cst, u16* __restrict__ H0b, u16* __restrict__ H1b,
            float* __restrict__ Out, unsigned* __restrict__ bar) {
  // [slice][gate][col][k]  = 2*4*16*1024*2B = 256 KB (<= 320 KB per WGP)
  __shared__ u16 lds[2][NGATE][16][HIDN];

  const int wv   = threadIdx.x >> 5;            // wave in block: 0..3
  const int wid  = blockIdx.x * 4 + wv;         // 0..127
  const int ls   = wv >> 1;                     // local slice 0..1
  const int m0   = (wid & 1) << 4;              // batch half: 0 or 16
  const int n0   = (wid >> 1) << 4;             // global hidden column base
  const int lane = threadIdx.x & 31;
  const int ln   = lane & 15;
  const int hi   = lane >> 4;
  const int koff = hi << 3;
  const int n    = n0 + ln;

  // ---- stage W_h slices: 128 rows (2 slices x 4 gates x 16 cols) x 1024 k
  {
    uint4* dst = (uint4*)&lds[0][0][0][0];
    for (int row = 0; row < 2 * NGATE * 16; ++row) {
      const int rsl = row >> 6;                 // slice in block
      const int rg  = (row >> 4) & 3;           // gate
      const int rc  = row & 15;                 // column in slice
      const int gn  = (blockIdx.x * 2 + rsl) * 16 + rc;
      const uint4* src = (const uint4*)(Wht + ((size_t)rg * HIDN + gn) * HIDN);
      if (threadIdx.x < 128)
        dst[row * 128 + threadIdx.x] = src[threadIdx.x];
    }
  }
  __syncthreads();

  const v8f z = {0.f, 0.f, 0.f, 0.f, 0.f, 0.f, 0.f, 0.f};

  for (int t = 0; t < SEQ; ++t) {
    const u16* Hin = (t & 1) ? H1b : H0b;
    u16*      Hout = (t & 1) ? H0b : H1b;

    // prefetch next step's Xpre slice (HBM -> cache) behind the GEMM
    if (t + 1 < SEQ) {
      const float4* xpn = (const float4*)Xpre + (size_t)(t + 1) * BATCH * HIDN;
#pragma unroll
      for (int r = 0; r < 8; ++r) {
        const int b = m0 + r + (hi << 3);
        __builtin_prefetch(&xpn[(size_t)b * HIDN + n], 0, 1);
      }
    }

    v8f ai = z, af = z, ag = z, ao = z;
    const u16* A = Hin + (size_t)m0 * HIDN;
#pragma unroll 2
    for (int k0 = 0; k0 < HIDN; k0 += 32) {
      v16bf a = load_frag(A + k0, HIDN);
      FragU bu;
      const u16* p;
      p = &lds[ls][0][ln][k0 + koff];
      bu.q[0] = *(const uint4*)p; bu.q[1] = *(const uint4*)(p + 16);
      ai = wmma_bf16(a, bu.v, ai);
      p = &lds[ls][1][ln][k0 + koff];
      bu.q[0] = *(const uint4*)p; bu.q[1] = *(const uint4*)(p + 16);
      af = wmma_bf16(a, bu.v, af);
      p = &lds[ls][2][ln][k0 + koff];
      bu.q[0] = *(const uint4*)p; bu.q[1] = *(const uint4*)(p + 16);
      ag = wmma_bf16(a, bu.v, ag);
      p = &lds[ls][3][ln][k0 + koff];
      bu.q[0] = *(const uint4*)p; bu.q[1] = *(const uint4*)(p + 16);
      ao = wmma_bf16(a, bu.v, ao);
    }

    const float4* xp = (const float4*)Xpre + (size_t)t * BATCH * HIDN;
#pragma unroll
    for (int r = 0; r < 8; ++r) {
      const int b = m0 + r + (hi << 3);
      const size_t idx = (size_t)b * HIDN + n;
      const float4 xv = xp[idx];
      const float iv = sigmoidf_(ai[r] + xv.x);
      const float fv = sigmoidf_(af[r] + xv.y);
      const float gv = tanhf_   (ag[r] + xv.z);
      const float ov = sigmoidf_(ao[r] + xv.w);
      const float cn = fv * Cst[idx] + iv * gv;
      const float hn = ov * tanhf_(cn);
      Cst[idx]  = cn;
      Hout[idx] = f2bf(hn);
      Out[((size_t)b * SEQ + t) * HIDN + n] = hn;
    }

    grid_sync(&bar[0], &bar[1], (unsigned)gridDim.x);
  }
}

__global__ void k_final(const float* __restrict__ out, const float* __restrict__ cst,
                        float* __restrict__ dht, float* __restrict__ dct) {
  int i = blockIdx.x * blockDim.x + threadIdx.x;
  if (i >= BATCH * HIDN) return;
  int b = i >> 10, h = i & 1023;
  dht[i] = out[((size_t)b * SEQ + (SEQ - 1)) * HIDN + h];
  dct[i] = cst[i];
}

// ------------------------------------------------------------------- launcher

extern "C" void kernel_launch(void* const* d_in, const int* in_sizes, int n_in,
                              void* d_out, int out_size, void* d_ws, size_t ws_size,
                              hipStream_t stream) {
  const float* x  = (const float*)d_in[0];
  const float* h0 = (const float*)d_in[1];
  const float* c0 = (const float*)d_in[2];
  const float* Wi[4] = {(const float*)d_in[3], (const float*)d_in[4],
                        (const float*)d_in[5], (const float*)d_in[6]};
  const float* Wh[4] = {(const float*)d_in[7], (const float*)d_in[8],
                        (const float*)d_in[9], (const float*)d_in[10]};
  const float* bii = (const float*)d_in[11];
  const float* bif = (const float*)d_in[12];
  const float* big = (const float*)d_in[13];
  const float* bio = (const float*)d_in[14];
  const float* bhi = (const float*)d_in[15];
  const float* bhf = (const float*)d_in[16];
  const float* bhg = (const float*)d_in[17];
  const float* bho = (const float*)d_in[18];

  char* p = (char*)d_ws;
  u16*   Xbf  = (u16*)p;      p += (size_t)SEQ * BATCH * IND * sizeof(u16);   // 32 MB
  u16*   Wt   = (u16*)p;      p += (size_t)NGATE * HIDN * IND * sizeof(u16);  //  8 MB
  u16*   Wht  = (u16*)p;      p += (size_t)NGATE * HIDN * HIDN * sizeof(u16); //  8 MB
  float* Bias = (float*)p;    p += (size_t)NGATE * HIDN * sizeof(float);
  float* Cst  = (float*)p;    p += (size_t)BATCH * HIDN * sizeof(float);
  u16*   H0b  = (u16*)p;      p += (size_t)BATCH * HIDN * sizeof(u16);
  u16*   H1b  = (u16*)p;      p += (size_t)BATCH * HIDN * sizeof(u16);
  unsigned* bar = (unsigned*)p; p += 256;                                     // barrier
  float* Xpre = (float*)p;                                                    // 256 MB

  // prep
  k_cvt_x<<<(SEQ * BATCH * IND + 255) / 256, 256, 0, stream>>>(x, Xbf,
                                                               SEQ * BATCH * IND);
  dim3 tb(32, 8), tg(32, 32);
  for (int g = 0; g < 4; ++g) {
    k_transpose_cvt<<<tg, tb, 0, stream>>>(Wi[g], Wt  + (size_t)g * HIDN * IND);
    k_transpose_cvt<<<tg, tb, 0, stream>>>(Wh[g], Wht + (size_t)g * HIDN * HIDN);
  }
  k_bias<<<16, 256, 0, stream>>>(bii, bif, big, bio, bhi, bhf, bhg, bho, Bias);
  k_init<<<(BATCH * HIDN + 255) / 256, 256, 0, stream>>>(h0, c0, H0b, Cst, bar);

  // phase 1: 65536 waves (512 m-tiles x 128 n-tiles), 8 waves/block
  k_gemm_x<<<8192, 256, 0, stream>>>(Xbf, Wt, Bias, Xpre);

  // phase 2: one persistent kernel runs the entire 512-step scan
  float* out = (float*)d_out;
  float* dht = out + (size_t)BATCH * SEQ * HIDN;
  float* dct = dht + (size_t)BATCH * HIDN;
  k_scan<<<SCAN_BLOCKS, SCAN_THREADS, 0, stream>>>(Wht, Xpre, Cst, H0b, H1b,
                                                   out, bar);

  k_final<<<(BATCH * HIDN + 255) / 256, 256, 0, stream>>>(out, Cst, dht, dct);
}
